// EFFM_33904471834942
// MI455X (gfx1250) — matrix operand order repaired and checked by reading, not compiled, not run on hardware
//
#include <hip/hip_runtime.h>
#include <hip/hip_bf16.h>
#include <stdint.h>

// ---------------------------------------------------------------------------
// EFFM on MI455X (gfx1250, wave32).
// Heavy ops (two 3x3 group convs + concat 1x1) run as f16 WMMA implicit GEMM
// with f32 accumulation (v_wmma_f32_16x16x32_f16), concat-conv fused into the
// conv epilogue through an LDS f16 restage + second WMMA stage.
// ---------------------------------------------------------------------------

typedef _Float16 f16;
typedef __attribute__((ext_vector_type(16))) _Float16 v16h;
typedef __attribute__((ext_vector_type(8)))  _Float16 v8h;
typedef __attribute__((ext_vector_type(8)))  float    v8f;

#define NG   128          // B * GROUPS
#define C    64           // channels per group
#define HH   64
#define WW   64
#define HWP  4096         // H*W
#define PW   68           // padded width/height (pad=2 each side covers dil<=2)

// ---------------- zero helper (pads of transposed buffer) -------------------
__global__ __launch_bounds__(256) void k_zero(uint32_t* p, size_t n) {
  size_t i = (size_t)blockIdx.x * blockDim.x + threadIdx.x;
  size_t s = (size_t)gridDim.x * blockDim.x;
  for (; i < n; i += s) p[i] = 0u;
}

// ---------------- weight pre-swizzle into WMMA A-operand layout -------------
// A (16x32 f16): lane L holds row M=L%16; halves v0..15 = K=(L/16)*16+0..15 (contiguous order used here; B matches).
// Stored as [t][s][mt][lane][16] contiguous.
__global__ __launch_bounds__(256)
void k_prep(const float* __restrict__ wb1, const float* __restrict__ wb2,
            const float* __restrict__ wcat,
            f16* __restrict__ wA1, f16* __restrict__ wA2, f16* __restrict__ wcA) {
  int i = blockIdx.x * 256 + threadIdx.x;
  if (i < 9 * 2 * 4 * 32 * 16) {                    // 36864 = conv A entries
    int v  = i & 15, lane = (i >> 4) & 31, mt = (i >> 9) & 3;
    int s  = (i >> 11) & 1, t = i >> 12;            // t = ky*3+kx
    int o  = mt * 16 + (lane & 15);
    int kk = (lane >> 4) * 16 + v;
    int ic = s * 32 + kk;
    int ky = t / 3, kx = t % 3;
    size_t wi = ((size_t)(o * 64 + ic) * 3 + ky) * 3 + kx;   // (O,I,3,3)
    wA1[i] = (f16)wb1[wi];
    wA2[i] = (f16)wb2[wi];
  }
  if (i < 2 * 2 * 4 * 32 * 16) {                    // 8192 = w_cat A entries
    int v  = i & 15, lane = (i >> 4) & 31, mt = (i >> 9) & 3;
    int s  = (i >> 11) & 1, which = i >> 12;
    int o  = mt * 16 + (lane & 15);
    int kk = (lane >> 4) * 16 + v;
    int ic = which * 64 + s * 32 + kk;
    wcA[i] = (f16)wcat[o * 128 + ic];               // (64,128,1,1)
  }
}

// ---------------- transpose x -> padded channel-contiguous f16 --------------
// XT[g][y+2][x+2][c] f16, zero borders (pad=2). One block per (g,y) row.
__global__ __launch_bounds__(256)
void k_transpose(const float* __restrict__ gx, f16* __restrict__ XT) {
  __shared__ float t[64 * 65];
  int y = blockIdx.x, g = blockIdx.y, tid = threadIdx.x;
  int x = tid & 63, cq = tid >> 6;                  // cq 0..3
  const float* src = gx + (size_t)g * 64 * HWP + y * 64;
  #pragma unroll
  for (int i = 0; i < 16; ++i) {
    int c = cq * 16 + i;
    t[c * 65 + x] = src[(size_t)c * HWP + x];
  }
  __syncthreads();
  int xx = tid >> 2, q = tid & 3;                   // 16 channels per thread
  f16* dst = XT + (((size_t)g * PW + y + 2) * PW + (xx + 2)) * 64 + q * 16;
  #pragma unroll
  for (int i = 0; i < 16; ++i) dst[i] = (f16)t[(q * 16 + i) * 65 + xx];
}

// ---------------- row / column means ---------------------------------------
__global__ __launch_bounds__(256)
void k_rowcol(const float* __restrict__ gx, float* __restrict__ xh,
              float* __restrict__ xw) {
  __shared__ float t[64 * 65];
  int gc = blockIdx.x, tid = threadIdx.x;
  const float* src = gx + (size_t)gc * HWP;
  #pragma unroll
  for (int i = 0; i < 16; ++i) {
    int p = i * 256 + tid;
    t[(p >> 6) * 65 + (p & 63)] = src[p];
  }
  __syncthreads();
  if (tid < 64) {
    float s = 0.f;
    #pragma unroll
    for (int w = 0; w < 64; ++w) s += t[tid * 65 + w];
    xh[(size_t)gc * 64 + tid] = s * (1.f / 64.f);
  } else if (tid < 128) {
    int w = tid - 64; float s = 0.f;
    #pragma unroll
    for (int h = 0; h < 64; ++h) s += t[h * 65 + w];
    xw[(size_t)gc * 64 + w] = s * (1.f / 64.f);
  }
}

// ---------------- hw 64x64 matmul + sigmoid --------------------------------
__global__ __launch_bounds__(256)
void k_hw(const float* __restrict__ xh, const float* __restrict__ xw,
          const float* __restrict__ whw, const float* __restrict__ bhw,
          float* __restrict__ sh, float* __restrict__ sw) {
  __shared__ float In[64 * 128];
  __shared__ float Wm[64 * 64];
  int g = blockIdx.x, tid = threadIdx.x;
  #pragma unroll
  for (int i = 0; i < 32; ++i) {
    int e = i * 256 + tid; int c = e >> 7, k = e & 127;
    In[e] = (k < 64) ? xh[((size_t)g * 64 + c) * 64 + k]
                     : xw[((size_t)g * 64 + c) * 64 + (k - 64)];
  }
  #pragma unroll
  for (int i = 0; i < 16; ++i) { int e = i * 256 + tid; Wm[e] = whw[e]; }
  __syncthreads();
  int k = tid & 127, ob = tid >> 7;                 // 2 blocks of 32 outputs
  float out[32];
  #pragma unroll
  for (int j = 0; j < 32; ++j) out[j] = bhw[ob * 32 + j];
  for (int c = 0; c < 64; ++c) {
    float in = In[c * 128 + k];
    #pragma unroll
    for (int j = 0; j < 32; ++j) out[j] += Wm[(ob * 32 + j) * 64 + c] * in;
  }
  #pragma unroll
  for (int j = 0; j < 32; ++j) {
    int o = ob * 32 + j;
    float sg = 1.f / (1.f + __expf(-out[j]));
    if (k < 64) sh[((size_t)g * 64 + o) * 64 + k] = sg;
    else        sw[((size_t)g * 64 + o) * 64 + (k - 64)] = sg;
  }
}

// ---------------- gate + GroupNorm -> x1 (into d_out) ----------------------
__global__ __launch_bounds__(256)
void k_gate_gn(const float* __restrict__ gx, const float* __restrict__ sh,
               const float* __restrict__ sw, const float* __restrict__ gnw,
               const float* __restrict__ gnb, float* __restrict__ x1out) {
  int gc = blockIdx.x, c = gc & 63, tid = threadIdx.x;
  const float* src = gx + (size_t)gc * HWP;
  const float* shp = sh + (size_t)gc * 64;
  const float* swp = sw + (size_t)gc * 64;
  float vals[16]; float s = 0.f, s2 = 0.f;
  #pragma unroll
  for (int i = 0; i < 16; ++i) {
    int p = i * 256 + tid;
    float v = src[p] * shp[p >> 6] * swp[p & 63];
    vals[i] = v; s += v; s2 += v * v;
  }
  #pragma unroll
  for (int off = 16; off; off >>= 1) {
    s  += __shfl_down(s,  off, 32);
    s2 += __shfl_down(s2, off, 32);
  }
  __shared__ float rs[16]; __shared__ float mv[2];
  int wv = tid >> 5, ln = tid & 31;
  if (ln == 0) { rs[wv] = s; rs[8 + wv] = s2; }
  __syncthreads();
  if (tid == 0) {
    float S = 0.f, S2 = 0.f;
    for (int i = 0; i < 8; ++i) { S += rs[i]; S2 += rs[8 + i]; }
    float mu = S * (1.f / HWP);
    float var = S2 * (1.f / HWP) - mu * mu;
    mv[0] = mu; mv[1] = rsqrtf(var + 1e-5f);
  }
  __syncthreads();
  float mu = mv[0], wsc = gnw[c] * mv[1], b = gnb[c];
  #pragma unroll
  for (int i = 0; i < 16; ++i) {
    int p = i * 256 + tid;
    x1out[(size_t)gc * HWP + p] = (vals[i] - mu) * wsc + b;
  }
}

// ---------------- WMMA conv (3x3, dilated) + PReLU + residual + cat-1x1 ----
// Block = one group x 128 pixels (2 rows). 8 waves, one 16-pixel column each,
// full 64 out-channels in 4 accumulators. Stage 2 re-stages br to LDS f16 and
// applies the matching half of w_cat with 8 more WMMAs, producing x2.
template <int DIL, int PASS>
__global__ __launch_bounds__(256)
void k_conv(const float* __restrict__ gx, const f16* __restrict__ XT,
            const f16* __restrict__ wA, const f16* __restrict__ wcA,
            const float* __restrict__ bconv, const float* __restrict__ alpha,
            const float* __restrict__ bcat, float* __restrict__ x2) {
  __shared__ f16 lds_br[128 * 72];                  // [local pixel][ch], pitch 72
  const int g  = blockIdx.y;
  const int pb = blockIdx.x;                        // 32 blocks of 128 pixels
  const int tid = threadIdx.x;
  const int wv = tid >> 5, ln = tid & 31;
  const int col = ln & 15, kh = ln >> 4;            // B: N = col, K-half = kh
  const int y  = pb * 2 + (wv >> 2);
  const int x0 = (wv & 3) * 16;
  const int lp = wv * 16 + col;                     // local pixel 0..127
  const int p  = y * 64 + x0 + col;                 // global pixel

  const f16* xg = XT + (size_t)g * (PW * PW * 64);
  v8f acc[4] = {};
  #pragma unroll
  for (int t = 0; t < 9; ++t) {
    const int ky = t / 3, kx = t % 3;
    const int sy = y + (ky - 1) * DIL + 2;
    const int sx = x0 + col + (kx - 1) * DIL + 2;   // pads are zero: no masking
    const f16* bp = xg + ((size_t)sy * PW + sx) * 64 + kh * 16;
    const f16* ap = wA + (size_t)(t * 2) * 2048 + ln * 16;
    #pragma unroll
    for (int s = 0; s < 2; ++s) {
      v16h b = *(const v16h*)(bp + s * 32);         // 16 contiguous channels
      #pragma unroll
      for (int mt = 0; mt < 4; ++mt) {
        v16h a = *(const v16h*)(ap + s * 2048 + mt * 512);
        acc[mt] = __builtin_amdgcn_wmma_f32_16x16x32_f16(
            false, a, false, b, (short)0, acc[mt], false, false);
      }
    }
  }
  // epilogue-1: bias + PReLU + residual, stash f16 tile to LDS
  const float al = alpha[0];
  #pragma unroll
  for (int mt = 0; mt < 4; ++mt)
    #pragma unroll
    for (int r = 0; r < 8; ++r) {
      int ch = mt * 16 + r + 8 * kh;                // C/D layout mapping
      float v = acc[mt][r] + bconv[ch];
      v = (v >= 0.f) ? v : al * v;
      v += gx[((size_t)g * 64 + ch) * HWP + p];     // residual in f32
      lds_br[lp * 72 + ch] = (f16)v;
    }
  __syncthreads();
  // stage-2: x2 (+)= Wcat_half @ br   (B operand straight from LDS)
  v8f a2[4] = {};
  #pragma unroll
  for (int s = 0; s < 2; ++s) {
    const f16* lbp = &lds_br[lp * 72 + s * 32 + kh * 16];
    v8h blo = *(const v8h*)(lbp);
    v8h bhi = *(const v8h*)(lbp + 8);
    v16h b = __builtin_shufflevector(blo, bhi, 0, 1, 2, 3, 4, 5, 6, 7,
                                     8, 9, 10, 11, 12, 13, 14, 15);
    #pragma unroll
    for (int mt = 0; mt < 4; ++mt) {
      v16h a = *(const v16h*)(wcA + s * 2048 + mt * 512 + ln * 16);
      a2[mt] = __builtin_amdgcn_wmma_f32_16x16x32_f16(
          false, a, false, b, (short)0, a2[mt], false, false);
    }
  }
  #pragma unroll
  for (int mt = 0; mt < 4; ++mt)
    #pragma unroll
    for (int r = 0; r < 8; ++r) {
      int o = mt * 16 + r + 8 * kh;
      size_t idx = ((size_t)g * 64 + o) * HWP + p;
      if (PASS == 0) x2[idx] = a2[mt][r] + bcat[o];
      else           x2[idx] += a2[mt][r];          // prior kernel finished
    }
}

// ---------------- x21 = softmax(mean_HW x2); x11 = softmax(gn_b) -----------
__global__ __launch_bounds__(256)
void k_x21(const float* __restrict__ x2, const float* __restrict__ gnb,
           float* __restrict__ x21, float* __restrict__ x11) {
  __shared__ float sp[64 * 8];
  __shared__ float m[64];
  int g = blockIdx.x, tid = threadIdx.x, wv = tid >> 5, ln = tid & 31;
  const float* src = x2 + (size_t)g * 64 * HWP;
  for (int c = 0; c < 64; ++c) {
    float pc = 0.f;
    #pragma unroll
    for (int i = 0; i < 16; ++i) pc += src[(size_t)c * HWP + i * 256 + tid];
    #pragma unroll
    for (int off = 16; off; off >>= 1) pc += __shfl_down(pc, off, 32);
    if (ln == 0) sp[c * 8 + wv] = pc;               // deterministic order
  }
  __syncthreads();
  if (tid == 0) {
    float mx = -1e30f;
    for (int c = 0; c < 64; ++c) {
      float s = 0.f;
      for (int i = 0; i < 8; ++i) s += sp[c * 8 + i];
      m[c] = s * (1.f / HWP);
      mx = fmaxf(mx, m[c]);
    }
    float ss = 0.f;
    for (int c = 0; c < 64; ++c) { m[c] = __expf(m[c] - mx); ss += m[c]; }
    float inv = 1.f / ss;
    for (int c = 0; c < 64; ++c) x21[(size_t)g * 64 + c] = m[c] * inv;
    if (g == 0) {                                   // mean(x1)==gn_b exactly
      float mb = -1e30f;
      for (int c = 0; c < 64; ++c) { m[c] = gnb[c]; mb = fmaxf(mb, m[c]); }
      float sb = 0.f;
      for (int c = 0; c < 64; ++c) { m[c] = __expf(m[c] - mb); sb += m[c]; }
      float ib = 1.f / sb;
      for (int c = 0; c < 64; ++c) x11[c] = m[c] * ib;
    }
  }
}

// ---------------- weights + final gating (x1 aliases out) ------------------
__global__ __launch_bounds__(256)
void k_final(const float* __restrict__ gx, const float* x1,
             const float* __restrict__ x2, const float* __restrict__ x11,
             const float* __restrict__ x21, float* out) {
  int g = blockIdx.y;
  int p = blockIdx.x * 256 + threadIdx.x;
  const float* a2 = x21 + (size_t)g * 64;
  size_t base = (size_t)g * 64 * HWP + p;
  float wsum = 0.f;
  for (int c = 0; c < 64; ++c)
    wsum += x11[c] * x2[base + (size_t)c * HWP] +
            a2[c]  * x1[base + (size_t)c * HWP];    // all reads before writes
  float sg = 1.f / (1.f + __expf(-wsum));
  for (int c = 0; c < 64; ++c) {
    size_t idx = base + (size_t)c * HWP;
    out[idx] = gx[idx] * sg;
  }
}

// ---------------------------------------------------------------------------
extern "C" void kernel_launch(void* const* d_in, const int* in_sizes, int n_in,
                              void* d_out, int out_size, void* d_ws, size_t ws_size,
                              hipStream_t stream) {
  (void)in_sizes; (void)n_in; (void)out_size; (void)ws_size;
  const float* x    = (const float*)d_in[0];
  const float* w_hw = (const float*)d_in[1];
  const float* b_hw = (const float*)d_in[2];
  const float* gn_w = (const float*)d_in[3];
  const float* gn_b = (const float*)d_in[4];
  const float* w_b1 = (const float*)d_in[5];
  const float* b_b1 = (const float*)d_in[6];
  const float* a_b1 = (const float*)d_in[7];
  const float* w_b2 = (const float*)d_in[8];
  const float* b_b2 = (const float*)d_in[9];
  const float* a_b2 = (const float*)d_in[10];
  const float* w_cat= (const float*)d_in[11];
  const float* b_cat= (const float*)d_in[12];

  char* ws = (char*)d_ws;
  const size_t szXT = (size_t)NG * PW * PW * 64 * 2;   // 75,759,616  (f16)
  const size_t szX2 = (size_t)NG * 64 * HWP * 4;       // 134,217,728 (f32)
  const size_t szV  = (size_t)NG * 64 * 64 * 4;        // 2,097,152
  const size_t szW  = (size_t)9 * 2 * 4 * 32 * 16 * 2; // 73,728
  const size_t szWC = (size_t)2 * 2 * 4 * 32 * 16 * 2; // 16,384
  size_t o = 0;
  f16*   XT  = (f16*)  (ws + o); o += szXT;
  float* x2f = (float*)(ws + o); o += szX2;
  float* xh  = (float*)(ws + o); o += szV;
  float* xw  = (float*)(ws + o); o += szV;
  float* sh  = (float*)(ws + o); o += szV;
  float* sw  = (float*)(ws + o); o += szV;
  f16*   wA1 = (f16*)  (ws + o); o += szW;
  f16*   wA2 = (f16*)  (ws + o); o += szW;
  f16*   wcA = (f16*)  (ws + o); o += szWC;
  float* x21 = (float*)(ws + o); o += (size_t)NG * 64 * 4;
  float* x11 = (float*)(ws + o); o += 256;

  float* out = (float*)d_out;                        // also hosts x1

  k_zero     <<<4096, 256, 0, stream>>>((uint32_t*)XT, szXT / 4);
  k_prep     <<<144,  256, 0, stream>>>(w_b1, w_b2, w_cat, wA1, wA2, wcA);
  k_transpose<<<dim3(64, NG), 256, 0, stream>>>(x, XT);
  k_rowcol   <<<NG * 64, 256, 0, stream>>>(x, xh, xw);
  k_hw       <<<NG, 256, 0, stream>>>(xh, xw, w_hw, b_hw, sh, sw);
  k_gate_gn  <<<NG * 64, 256, 0, stream>>>(x, sh, sw, gn_w, gn_b, out);
  k_conv<1, 0><<<dim3(32, NG), 256, 0, stream>>>(x, XT, wA1, wcA,
                                                 b_b1, a_b1, b_cat, x2f);
  k_conv<2, 1><<<dim3(32, NG), 256, 0, stream>>>(x, XT, wA2, wcA + 2 * 2048,
                                                 b_b2, a_b2, b_cat, x2f);
  k_x21      <<<NG, 256, 0, stream>>>(x2f, gn_b, x21, x11);
  k_final    <<<dim3(16, NG), 256, 0, stream>>>(x, out, x2f, x11, x21, out);
}
